// LSTM_40381282517505
// MI455X (gfx1250) — matrix operand order repaired and checked by reading, not compiled, not run on hardware
//
#include <hip/hip_runtime.h>
#include <hip/hip_bf16.h>
#include <stdint.h>

// ---------------------------------------------------------------------------
// Problem constants (from reference)
// ---------------------------------------------------------------------------
#define N_FRAMES 750
#define HIDDEN   512
#define INPUT    188
#define NCLS     56
#define BATCH    64
#define FEATS    (N_FRAMES * HIDDEN)   // 384000
#define LINW_STRIDE (FEATS + 1)        // 384001 (row stride of lin_w)

// Workspace layout (bytes)
#define WS_CNT_OFF  0                         // 4 group counters, 64B apart
#define WS_HBUF_OFF 1024                      // ping-pong h: 2*4*16*512 bf16 = 131072 B
#define WS_HIST_OFF (1024 + 131072)           // h history: 64*384000 bf16 = 49.15 MB

// LSTM kernel dynamic LDS carve (bytes)
#define LDS_WHH   131072   // 16 ktiles * 8 ntiles * (32 lanes * 16 bf16)
#define LDS_WIH   49152    // 6 ktiles  * 8 ntiles * 1KB
#define LDS_HFRAG 16384    // 16 ktiles * 1KB
#define LDS_XFRAG 6144     // 6 ktiles  * 1KB
#define LDS_GATE  8192     // 16 m * 128 gates * f32
#define LDS_CBUF  2048     // 16 m * 32 cols * f32
#define LDS_BIAS  512      // 128 * f32
#define LSTM_LDS_BYTES (LDS_WHH + LDS_WIH + LDS_HFRAG + LDS_XFRAG + LDS_GATE + LDS_CBUF + LDS_BIAS)

typedef __attribute__((ext_vector_type(16))) __bf16 v16bf;
typedef __attribute__((ext_vector_type(8)))  float  v8f;

union FragB { uint4 q[2]; v16bf v; };  // 32 bytes = 8 VGPRs/lane

// ---- CDNA5 async global->LDS path (guarded; falls back to VGPR bounce) ----
#if defined(__has_builtin)
#  if __has_builtin(__builtin_amdgcn_global_load_async_to_lds_b128)
#    define HAVE_ASYNC_G2L 1
#  endif
#endif
#ifndef HAVE_ASYNC_G2L
#  define HAVE_ASYNC_G2L 0
#endif

#if HAVE_ASYNC_G2L
typedef int v4i_t __attribute__((vector_size(16)));
typedef __attribute__((address_space(1))) v4i_t* as1_v4i_p;   // global
typedef __attribute__((address_space(3))) v4i_t* as3_v4i_p;   // LDS

__device__ __forceinline__ void async_g2l_b128(void* lds, const void* gptr) {
  __builtin_amdgcn_global_load_async_to_lds_b128(
      (as1_v4i_p)(gptr), (as3_v4i_p)(lds), 0, 0);
}
__device__ __forceinline__ void wait_async0() {
#if defined(__has_builtin) && __has_builtin(__builtin_amdgcn_s_wait_asynccnt)
  __builtin_amdgcn_s_wait_asynccnt(0);
#else
  asm volatile("s_wait_asynccnt 0x0" ::: "memory");
#endif
}
#endif

__device__ __forceinline__ uint16_t f2bf(float f) {
  uint32_t u = __float_as_uint(f);
  u = (u + 0x7FFFu + ((u >> 16) & 1u)) >> 16;   // round-to-nearest-even
  return (uint16_t)u;
}
__device__ __forceinline__ float sigf(float x) {
  return __builtin_amdgcn_rcpf(1.0f + __expf(-x));   // v_exp_f32 + v_rcp_f32
}
__device__ __forceinline__ float tanhx(float x) {
  return 2.0f * sigf(2.0f * x) - 1.0f;
}

// ---------------------------------------------------------------------------
// Persistent LSTM kernel.
// Grid = 64 blocks x 256 threads (8 wave32).
//   group = blockIdx.x>>4 : 16 batch rows  (independent; own barrier counter)
//   slice = blockIdx.x&15 : 32 h-columns -> 128 gate rows {i,f,g,o}x32
// w_hh/w_ih slices live in LDS (bf16, WMMA-B fragment order) for all 750 steps.
// Per step: stage h (16KB via L2, async-to-LDS) + x_t into A-fragment order,
// 22 bf16 WMMAs per wave (16 over h, 6 over padded-192 input), pointwise cell
// in LDS, publish new h slice + history, signal group barrier.
// ---------------------------------------------------------------------------
__global__ __launch_bounds__(256) void lstm_kernel(
    const float* __restrict__ local_f, const float* __restrict__ w_ih,
    const float* __restrict__ w_hh, const float* __restrict__ b_ih,
    const float* __restrict__ b_hh, uint8_t* __restrict__ ws) {
  extern __shared__ char smem[];
  char*  whh_frag = smem;
  char*  wih_frag = whh_frag + LDS_WHH;
  char*  h_frag   = wih_frag + LDS_WIH;
  char*  x_frag   = h_frag + LDS_HFRAG;
  float* gate_buf = (float*)(x_frag + LDS_XFRAG);
  float* c_buf    = (float*)((char*)gate_buf + LDS_GATE);
  float* bias     = (float*)((char*)c_buf + LDS_CBUF);

  const int tid   = threadIdx.x;
  const int lane  = tid & 31;
  const int wv    = tid >> 5;
  const int slice = blockIdx.x & 15;
  const int group = blockIdx.x >> 4;

  // ---- one-time: convert weight slices to bf16 fragments in LDS ----
  // B-matrix 32x16 bf16 layout: lane -> N=lane&15; dwords 0..3: K=(lane>>4)*8+2v{+1}, dwords 4..7: +16
  for (int idx = tid; idx < 16 * 8 * 32 * 8; idx += 256) {
    int v = idx & 7, ln = (idx >> 3) & 31, blk = idx >> 8;
    int nt = blk & 7, kt = blk >> 3;
    int hf = ln >> 4;
    int n_l = nt * 16 + (ln & 15);                 // local gate row 0..127
    int K = kt * 32 + ((v & 4) ? 16 : 0) + hf * 8 + (v & 3) * 2;
    int gate = n_l >> 5;
    int grow = gate * HIDDEN + slice * 32 + (n_l & 31);
    const float* src = w_hh + (size_t)grow * HIDDEN + K;
    ((uint32_t*)whh_frag)[idx] =
        (uint32_t)f2bf(src[0]) | ((uint32_t)f2bf(src[1]) << 16);
  }
  for (int idx = tid; idx < 6 * 8 * 32 * 8; idx += 256) {
    int v = idx & 7, ln = (idx >> 3) & 31, blk = idx >> 8;
    int nt = blk & 7, kt = blk >> 3;
    int hf = ln >> 4;
    int n_l = nt * 16 + (ln & 15);
    int K = kt * 32 + ((v & 4) ? 16 : 0) + hf * 8 + (v & 3) * 2;
    int gate = n_l >> 5;
    int grow = gate * HIDDEN + slice * 32 + (n_l & 31);
    float f0 = (K     < INPUT) ? w_ih[(size_t)grow * INPUT + K]     : 0.f;
    float f1 = (K + 1 < INPUT) ? w_ih[(size_t)grow * INPUT + K + 1] : 0.f;
    ((uint32_t*)wih_frag)[idx] =
        (uint32_t)f2bf(f0) | ((uint32_t)f2bf(f1) << 16);
  }
  if (tid < 128) {
    int gate = tid >> 5;
    int grow = gate * HIDDEN + slice * 32 + (tid & 31);
    bias[tid] = b_ih[grow] + b_hh[grow];
  }
  c_buf[tid] = 0.f;
  c_buf[tid + 256] = 0.f;
  __syncthreads();

  uint32_t* cnt  = (uint32_t*)(ws + WS_CNT_OFF) + group * 16;
  uint16_t* hbuf = (uint16_t*)(ws + WS_HBUF_OFF);
  uint16_t* hist = (uint16_t*)(ws + WS_HIST_OFF);

  for (int t = 0; t < N_FRAMES; ++t) {
    // ---- group barrier: wait for all 16 WGs to finish step t-1 ----
    if (t > 0) {
      if (tid == 0) {
        const uint32_t target = (uint32_t)(16 * t);
        while (__hip_atomic_load(cnt, __ATOMIC_ACQUIRE,
                                 __HIP_MEMORY_SCOPE_AGENT) < target)
          __builtin_amdgcn_s_sleep(2);
      }
      __syncthreads();
    }

    // ---- stage h into A-fragment layout (parity (t+1)&1; zeroed for t=0) ----
    // A 16x32 bf16: lane -> M=lane&15; dwords 0..3: K=hf*8+2v{+1}, dwords 4..7: +16
    const uint16_t* hsrc =
        hbuf + ((size_t)(((t + 1) & 1) * 4 + group)) * 16 * HIDDEN;
    {
      int pr = tid;
      for (int r = 0; r < 2; ++r, pr += 256) {   // 512 (ktile,lane) pairs
        int kt = pr >> 5, ln = pr & 31;
        int hf = ln >> 4, m = ln & 15;
        const uint16_t* row = hsrc + (size_t)m * HIDDEN + kt * 32 + hf * 8;
        char* dst = h_frag + kt * 1024 + ln * 32;
#if HAVE_ASYNC_G2L
        async_g2l_b128(dst, row);            // global_load_async_to_lds_b128
        async_g2l_b128(dst + 16, row + 16);
#else
        uint4 a0 = *(const uint4*)row;
        uint4 a1 = *(const uint4*)(row + 16);
        ((uint4*)dst)[0] = a0;
        ((uint4*)dst)[1] = a1;
#endif
      }
    }
    // ---- stage x_t (K padded to 192; needs f32->bf16, stays on VGPR path) ----
    if (tid < 192) {
      int kt = tid >> 5, ln = tid & 31;
      int hf = ln >> 4, m = ln & 15;
      const float* xrow =
          local_f + ((size_t)(group * 16 + m)) * N_FRAMES * INPUT +
          (size_t)t * INPUT;
      uint32_t dw[8];
#pragma unroll
      for (int v = 0; v < 8; ++v) {
        int K = kt * 32 + ((v & 4) ? 16 : 0) + hf * 8 + (v & 3) * 2;
        float f0 = (K     < INPUT) ? xrow[K]     : 0.f;
        float f1 = (K + 1 < INPUT) ? xrow[K + 1] : 0.f;
        dw[v] = (uint32_t)f2bf(f0) | ((uint32_t)f2bf(f1) << 16);
      }
      uint4* dst = (uint4*)(x_frag + kt * 1024 + ln * 32);
      dst[0] = make_uint4(dw[0], dw[1], dw[2], dw[3]);
      dst[1] = make_uint4(dw[4], dw[5], dw[6], dw[7]);
    }
#if HAVE_ASYNC_G2L
    wait_async0();
#endif
    __syncthreads();

    // ---- WMMA: wave wv owns gate rows [wv*16, wv*16+16), all 16 batch rows ----
    v8f acc = {0.f, 0.f, 0.f, 0.f, 0.f, 0.f, 0.f, 0.f};
#pragma unroll
    for (int kt = 0; kt < 16; ++kt) {
      FragB a, b;
      const uint4* ap = (const uint4*)(h_frag + kt * 1024 + lane * 32);
      a.q[0] = ap[0]; a.q[1] = ap[1];
      const uint4* bp = (const uint4*)(whh_frag + (kt * 8 + wv) * 1024 + lane * 32);
      b.q[0] = bp[0]; b.q[1] = bp[1];
      acc = __builtin_amdgcn_wmma_f32_16x16x32_bf16(
          false, a.v, false, b.v, (short)0, acc, false, false);
    }
#pragma unroll
    for (int kt = 0; kt < 6; ++kt) {
      FragB a, b;
      const uint4* ap = (const uint4*)(x_frag + kt * 1024 + lane * 32);
      a.q[0] = ap[0]; a.q[1] = ap[1];
      const uint4* bp = (const uint4*)(wih_frag + (kt * 8 + wv) * 1024 + lane * 32);
      b.q[0] = bp[0]; b.q[1] = bp[1];
      acc = __builtin_amdgcn_wmma_f32_16x16x32_bf16(
          false, a.v, false, b.v, (short)0, acc, false, false);
    }
    // C/D layout: M = v + 8*(lane>>4), N = lane&15
    {
      int n_l = wv * 16 + (lane & 15);
      int hf = lane >> 4;
      float bb = bias[n_l];
#pragma unroll
      for (int v = 0; v < 8; ++v) {
        int m = v + 8 * hf;
        gate_buf[m * 128 + n_l] = acc[v] + bb;
      }
    }
    __syncthreads();

    // ---- pointwise LSTM cell (i,f,g,o) + publish h ----
    {
      int e = tid;
      for (int r = 0; r < 2; ++r, e += 256) {   // 16 m * 32 cols
        int m = e >> 5, col = e & 31;
        float gi = sigf(gate_buf[m * 128 + col]);
        float gf = sigf(gate_buf[m * 128 + 32 + col]);
        float gg = tanhx(gate_buf[m * 128 + 64 + col]);
        float go = sigf(gate_buf[m * 128 + 96 + col]);
        float c = gf * c_buf[e] + gi * gg;
        c_buf[e] = c;
        float h = go * tanhx(c);
        uint16_t hb = f2bf(h);
        hbuf[((size_t)((t & 1) * 4 + group)) * 16 * HIDDEN +
             (size_t)m * HIDDEN + slice * 32 + col] = hb;
        hist[(size_t)(group * 16 + m) * FEATS + (size_t)t * HIDDEN +
             slice * 32 + col] = hb;
      }
    }
    __threadfence();
    __syncthreads();
    if (tid == 0)
      __hip_atomic_fetch_add(cnt, 1u, __ATOMIC_RELEASE,
                             __HIP_MEMORY_SCOPE_AGENT);
  }
}

// ---------------------------------------------------------------------------
// Head init: out[b,k] = lin_b[k] + global_f[b] * lin_w[k, 384000]
// ---------------------------------------------------------------------------
__global__ __launch_bounds__(256) void head_init_kernel(
    const float* __restrict__ global_f, const float* __restrict__ lin_w,
    const float* __restrict__ lin_b, float* __restrict__ out) {
  for (int idx = threadIdx.x; idx < BATCH * NCLS; idx += 256) {
    int b = idx / NCLS, k = idx - b * NCLS;
    out[idx] = lin_b[k] + global_f[b] * lin_w[(size_t)k * LINW_STRIDE + FEATS];
  }
}

// ---------------------------------------------------------------------------
// Head GEMM: out += hist(bf16)[64 x 384000] @ lin_w.T  via WMMA.
// 400 blocks, each reduces a 960-feature K-slab (30 ktiles of 32),
// converting lin_w tiles fp32->bf16 in LDS, atomic f32 reduction to d_out.
// ---------------------------------------------------------------------------
#define KCHUNK 960
#define KTILES 30
__global__ __launch_bounds__(256) void head_gemm_kernel(
    const float* __restrict__ lin_w, const uint8_t* __restrict__ ws,
    float* __restrict__ out) {
  __shared__ char a_frag[4 * 1024];   // 4 m-tiles (batch 64)
  __shared__ char b_frag[4 * 1024];   // 4 n-tiles (classes 56 padded to 64)
  const int tid = threadIdx.x, lane = tid & 31, wv = tid >> 5;
  const uint16_t* hist = (const uint16_t*)(ws + WS_HIST_OFF);
  const size_t kbase = (size_t)blockIdx.x * KCHUNK;

  const int mt1 = wv & 3,       nt1 = wv >> 2;        // tile wv
  const int mt2 = (wv + 8) & 3, nt2 = (wv + 8) >> 2;  // tile wv+8
  v8f acc1 = {0.f, 0.f, 0.f, 0.f, 0.f, 0.f, 0.f, 0.f};
  v8f acc2 = {0.f, 0.f, 0.f, 0.f, 0.f, 0.f, 0.f, 0.f};

  for (int kt = 0; kt < KTILES; ++kt) {
    const size_t kb = kbase + (size_t)kt * 32;
    if (tid < 128) {           // stage A: bf16 history, fragment order
      int mt = tid >> 5, ln = tid & 31;
      int hf = ln >> 4, m = mt * 16 + (ln & 15);
      const uint16_t* row = hist + (size_t)m * FEATS + kb + hf * 8;
      char* dst = a_frag + mt * 1024 + ln * 32;
#if HAVE_ASYNC_G2L
      async_g2l_b128(dst, row);
      async_g2l_b128(dst + 16, row + 16);
#else
      uint4 a0 = *(const uint4*)row;
      uint4 a1 = *(const uint4*)(row + 16);
      ((uint4*)dst)[0] = a0;
      ((uint4*)dst)[1] = a1;
#endif
    } else {                   // stage B: lin_w fp32 -> bf16, fragment order
      int q = tid - 128;
      int nt = q >> 5, ln = q & 31;
      int hf = ln >> 4, n_g = nt * 16 + (ln & 15);
      uint32_t dw[8];
      if (n_g < NCLS) {
        const float* src = lin_w + (size_t)n_g * LINW_STRIDE + kb + hf * 8;
#pragma unroll
        for (int v = 0; v < 8; ++v)
          dw[v] = (uint32_t)f2bf(src[2 * v + (v & 4) * 2]) |
                  ((uint32_t)f2bf(src[2 * v + (v & 4) * 2 + 1]) << 16);
        if (kt + 1 < KTILES) __builtin_prefetch(src + 32, 0, 1);
      } else {
#pragma unroll
        for (int v = 0; v < 8; ++v) dw[v] = 0u;
      }
      uint4* dst = (uint4*)(b_frag + nt * 1024 + ln * 32);
      dst[0] = make_uint4(dw[0], dw[1], dw[2], dw[3]);
      dst[1] = make_uint4(dw[4], dw[5], dw[6], dw[7]);
    }
#if HAVE_ASYNC_G2L
    wait_async0();
#endif
    __syncthreads();
    {
      FragB a, b;
      const uint4* ap = (const uint4*)(a_frag + mt1 * 1024 + lane * 32);
      a.q[0] = ap[0]; a.q[1] = ap[1];
      const uint4* bp = (const uint4*)(b_frag + nt1 * 1024 + lane * 32);
      b.q[0] = bp[0]; b.q[1] = bp[1];
      acc1 = __builtin_amdgcn_wmma_f32_16x16x32_bf16(
          false, a.v, false, b.v, (short)0, acc1, false, false);
      ap = (const uint4*)(a_frag + mt2 * 1024 + lane * 32);
      a.q[0] = ap[0]; a.q[1] = ap[1];
      bp = (const uint4*)(b_frag + nt2 * 1024 + lane * 32);
      b.q[0] = bp[0]; b.q[1] = bp[1];
      acc2 = __builtin_amdgcn_wmma_f32_16x16x32_bf16(
          false, a.v, false, b.v, (short)0, acc2, false, false);
    }
    __syncthreads();
  }

  const int hf = lane >> 4;
  {
    int n_g = nt1 * 16 + (lane & 15);
    if (n_g < NCLS) {
#pragma unroll
      for (int v = 0; v < 8; ++v)
        atomicAdd(out + (mt1 * 16 + v + 8 * hf) * NCLS + n_g, acc1[v]);
    }
  }
  {
    int n_g = nt2 * 16 + (lane & 15);
    if (n_g < NCLS) {
#pragma unroll
      for (int v = 0; v < 8; ++v)
        atomicAdd(out + (mt2 * 16 + v + 8 * hf) * NCLS + n_g, acc2[v]);
    }
  }
}

// ---------------------------------------------------------------------------
extern "C" void kernel_launch(void* const* d_in, const int* in_sizes, int n_in,
                              void* d_out, int out_size, void* d_ws,
                              size_t ws_size, hipStream_t stream) {
  const float* local_f  = (const float*)d_in[0];
  const float* global_f = (const float*)d_in[1];
  const float* w_ih     = (const float*)d_in[2];
  const float* w_hh     = (const float*)d_in[3];
  const float* b_ih     = (const float*)d_in[4];
  const float* b_hh     = (const float*)d_in[5];
  const float* lin_w    = (const float*)d_in[6];
  const float* lin_b    = (const float*)d_in[7];
  float*   out = (float*)d_out;
  uint8_t* ws  = (uint8_t*)d_ws;

  (void)in_sizes; (void)n_in; (void)out_size; (void)ws_size;

  (void)hipFuncSetAttribute(reinterpret_cast<const void*>(lstm_kernel),
                            hipFuncAttributeMaxDynamicSharedMemorySize,
                            LSTM_LDS_BYTES);

  // zero barrier counters + h ping-pong (h0 = 0); hist fully overwritten
  (void)hipMemsetAsync(ws, 0, WS_HIST_OFF, stream);

  lstm_kernel<<<64, 256, LSTM_LDS_BYTES, stream>>>(local_f, w_ih, w_hh, b_ih,
                                                   b_hh, ws);
  head_init_kernel<<<1, 256, 0, stream>>>(global_f, lin_w, lin_b, out);
  head_gemm_kernel<<<FEATS / KCHUNK, 256, 0, stream>>>(lin_w, ws, out);
}
